// GemmaAttention_73529840108084
// MI455X (gfx1250) — matrix operand order repaired and checked by reading, not compiled, not run on hardware
//
#include <hip/hip_runtime.h>
#include <hip/hip_bf16.h>

// ---------------------------------------------------------------------------
// GemmaAttention on MI455X (gfx1250, wave32).
// All matmul work (QKV proj, QK^T, PV, out proj) runs on v_wmma_f32_16x16x32_bf16.
// fp32->bf16 conversion is hoisted into a one-shot prep kernel so the GEMM
// inner loops are pure {vector load, WMMA}.
// ---------------------------------------------------------------------------

typedef __attribute__((ext_vector_type(16))) __bf16 v16bf;
typedef __attribute__((ext_vector_type(8)))  __bf16 v8bf;
typedef __attribute__((ext_vector_type(8)))  float  v8f;
typedef __attribute__((ext_vector_type(4)))  float  v4f;

#define T_TOK    2048
#define HIDDEN   2048
#define NUM_HEADS 8
#define HEAD_DIM 256
#define Q_SIZE   2048
#define KV_SIZE  256
#define QKV_N    2560   // Q_SIZE + 2*KV_SIZE

static __device__ __forceinline__ v8f wmma_bf16(v16bf a, v16bf b, v8f c) {
    // D = A(16x32 bf16) * B(32x16 bf16) + C(16x16 f32)
    return __builtin_amdgcn_wmma_f32_16x16x32_bf16(
        /*neg_a=*/false, a, /*neg_b=*/false, b,
        /*c_mod=*/(short)0, c, /*reuse_a=*/false, /*reuse_b=*/false);
}

static __device__ __forceinline__ v16bf cat8(v8bf lo, v8bf hi) {
    return __builtin_shufflevector(lo, hi, 0, 1, 2, 3, 4, 5, 6, 7,
                                           8, 9, 10, 11, 12, 13, 14, 15);
}

// ---------------------------------------------------------------------------
// fp32 -> bf16 bulk convert. n must be a multiple of 2048; 8 elems/thread.
// ---------------------------------------------------------------------------
__global__ __launch_bounds__(256, 4)
void cvt_bf16_kernel(const float* __restrict__ src, __bf16* __restrict__ dst)
{
    const size_t i = ((size_t)blockIdx.x * 256 + threadIdx.x) * 8;
    v4f x0 = *(const v4f*)(src + i);
    v4f x1 = *(const v4f*)(src + i + 4);
    v8bf y;
#pragma unroll
    for (int j = 0; j < 4; ++j) {
        y[j]     = (__bf16)x0[j];
        y[4 + j] = (__bf16)x1[j];
    }
    *(v8bf*)(dst + i) = y;
}

// ---------------------------------------------------------------------------
// C[M,N] (fp32) = A[M,K] * B[N,K]^T, A/B bf16 row-major.
// block = 256 threads = 8 waves; each wave computes a 32x32 tile.
// grid = (N/32, M/256)
// ---------------------------------------------------------------------------
__global__ __launch_bounds__(256, 1)
void gemm_nt_kernel(const __bf16* __restrict__ A, const __bf16* __restrict__ B,
                    float* __restrict__ C, int M, int N, int K)
{
    const int lane = threadIdx.x & 31;
    const int wave = threadIdx.x >> 5;
    const int r    = lane & 15;   // row (A) / col (B,C) within 16
    const int hi   = lane >> 4;   // half-wave selector
    const int n0   = blockIdx.x * 32;
    const int m0   = blockIdx.y * 256 + wave * 32;

    const __bf16* arow0 = A + (size_t)(m0 + r) * K;        // mt=0 row
    const __bf16* arow1 = A + (size_t)(m0 + 16 + r) * K;   // mt=1 row
    const __bf16* brow0 = B + (size_t)(n0 + r) * K;        // nt=0 col
    const __bf16* brow1 = B + (size_t)(n0 + 16 + r) * K;   // nt=1 col

    v8f acc[2][2] = {};

#pragma unroll 2
    for (int k0 = 0; k0 < K; k0 += 32) {
        // A fragment: elem j -> k = k0 + (j/8)*16 + (j%8) + hi*8
        const int ka = k0 + hi * 8;
        v16bf a0 = cat8(*(const v8bf*)(arow0 + ka), *(const v8bf*)(arow0 + ka + 16));
        v16bf a1 = cat8(*(const v8bf*)(arow1 + ka), *(const v8bf*)(arow1 + ka + 16));
        // B fragment: elem j -> k = k0 + (j/8)*16 + (j%8) + hi*8
        const int kbx = k0 + hi * 8;
        v16bf b0 = cat8(*(const v8bf*)(brow0 + kbx), *(const v8bf*)(brow0 + kbx + 16));
        v16bf b1 = cat8(*(const v8bf*)(brow1 + kbx), *(const v8bf*)(brow1 + kbx + 16));

        if (k0 + 32 < K) {  // global_prefetch_b8 of next K slab
            __builtin_prefetch(arow0 + k0 + 32, 0, 1);
            __builtin_prefetch(brow0 + k0 + 32, 0, 1);
        }

        acc[0][0] = wmma_bf16(a0, b0, acc[0][0]);
        acc[0][1] = wmma_bf16(a0, b1, acc[0][1]);
        acc[1][0] = wmma_bf16(a1, b0, acc[1][0]);
        acc[1][1] = wmma_bf16(a1, b1, acc[1][1]);
    }

    // C/D layout: elem j -> row m0+mt*16 + j + 8*hi, col n0+nt*16 + r
#pragma unroll
    for (int mt = 0; mt < 2; ++mt)
#pragma unroll
        for (int nt = 0; nt < 2; ++nt)
#pragma unroll
            for (int j = 0; j < 8; ++j)
                C[(size_t)(m0 + mt * 16 + hi * 8 + j) * N + n0 + nt * 16 + r] =
                    acc[mt][nt][j];
}

// ---------------------------------------------------------------------------
// RoPE (NeoX, rotate-halves) on Q and K; pack bf16 Q, bf16 K, bf16 V^T.
// One block (256 thr) per token.
// ---------------------------------------------------------------------------
__global__ __launch_bounds__(256, 1)
void rope_pack_kernel(const int* __restrict__ positions,
                      const float* __restrict__ qkv,
                      __bf16* __restrict__ qb,
                      __bf16* __restrict__ kb,
                      __bf16* __restrict__ vT)
{
    const int   t   = blockIdx.x;
    const float pos = (float)positions[t];
    const float* row = qkv + (size_t)t * QKV_N;
    const float lg = 0.0719558f;  // ln(10000)/128

    for (int idx = threadIdx.x; idx < (NUM_HEADS + 1) * 128; idx += 256) {
        const int h = idx >> 7;      // 0..7 = q heads, 8 = k head
        const int i = idx & 127;
        const float ang = pos * __expf(-lg * (float)i);
        float sn, cs;
        __sincosf(ang, &sn, &cs);
        if (h < NUM_HEADS) {
            const int base = h * HEAD_DIM;
            const float x1 = row[base + i], x2 = row[base + i + 128];
            qb[(size_t)t * Q_SIZE + base + i]       = (__bf16)(x1 * cs - x2 * sn);
            qb[(size_t)t * Q_SIZE + base + i + 128] = (__bf16)(x2 * cs + x1 * sn);
        } else {
            const float x1 = row[Q_SIZE + i], x2 = row[Q_SIZE + i + 128];
            kb[(size_t)t * KV_SIZE + i]       = (__bf16)(x1 * cs - x2 * sn);
            kb[(size_t)t * KV_SIZE + i + 128] = (__bf16)(x2 * cs + x1 * sn);
        }
    }
    // V transposed (d-major) so PV WMMA B-fragments are contiguous over keys.
    for (int i = threadIdx.x; i < KV_SIZE; i += 256)
        vT[(size_t)i * T_TOK + t] = (__bf16)row[Q_SIZE + KV_SIZE + i];
}

// ---------------------------------------------------------------------------
// Causal flash attention. block = 128 thr = 4 waves; each wave owns 16 query
// rows of one head; key blocks of 32. grid = (T/64, NUM_HEADS).
// Output written directly as bf16 (feeds the bf16 out-projection GEMM).
// ---------------------------------------------------------------------------
__global__ __launch_bounds__(128, 1)
void attn_fa_kernel(const __bf16* __restrict__ qb,
                    const __bf16* __restrict__ kb,
                    const __bf16* __restrict__ vT,
                    __bf16* __restrict__ attn)
{
    __shared__ __align__(32) __bf16 Pld[4][16][32];

    const int lane = threadIdx.x & 31;
    const int wave = threadIdx.x >> 5;
    const int r    = lane & 15;
    const int hi   = lane >> 4;
    const int head = blockIdx.y;
    const int qw0  = blockIdx.x * 64 + wave * 16;
    const float scale = 0.0625f;  // 1/sqrt(256)

    // Preload the wave's 16x256 Q block as 8 A-fragments.
    v16bf qf[8];
#pragma unroll
    for (int df = 0; df < 8; ++df) {
        const __bf16* qp = qb + (size_t)(qw0 + r) * Q_SIZE + head * HEAD_DIM
                              + df * 32 + hi * 8;
        qf[df] = cat8(*(const v8bf*)(qp), *(const v8bf*)(qp + 16));
    }

    v8f   o[16] = {};           // 16 rows x 256 d
    float mrow[8], lrow[8];
#pragma unroll
    for (int j = 0; j < 8; ++j) { mrow[j] = -3.0e38f; lrow[j] = 0.0f; }

    for (int kb0 = 0; kb0 < qw0 + 16; kb0 += 32) {
        // ---- S = Q K^T (16x32 scores; two 16x16 C tiles) ----
        v8f s[2] = {};
#pragma unroll
        for (int df = 0; df < 8; ++df) {
#pragma unroll
            for (int nt = 0; nt < 2; ++nt) {
                const __bf16* kp = kb + (size_t)(kb0 + nt * 16 + r) * KV_SIZE
                                      + df * 32 + hi * 8;
                v16bf kf = cat8(*(const v8bf*)(kp), *(const v8bf*)(kp + 16));
                s[nt] = wmma_bf16(qf[df], kf, s[nt]);
            }
        }
        // ---- scale + causal mask ----
#pragma unroll
        for (int nt = 0; nt < 2; ++nt)
#pragma unroll
            for (int j = 0; j < 8; ++j) {
                float v = s[nt][j] * scale;
                const int key = kb0 + nt * 16 + r;
                const int qr  = qw0 + j + 8 * hi;
                if (key > qr) v = -3.0e38f;
                s[nt][j] = v;
            }
        // ---- online softmax stats (rows live in 16-lane groups) ----
        float alpha[8];
#pragma unroll
        for (int j = 0; j < 8; ++j) {
            float v = fmaxf(s[0][j], s[1][j]);
            v = fmaxf(v, __shfl_xor(v, 1, 32));
            v = fmaxf(v, __shfl_xor(v, 2, 32));
            v = fmaxf(v, __shfl_xor(v, 4, 32));
            v = fmaxf(v, __shfl_xor(v, 8, 32));
            const float mnew = fmaxf(mrow[j], v);
            alpha[j] = __expf(mrow[j] - mnew);
            mrow[j]  = mnew;
        }
#pragma unroll
        for (int j = 0; j < 8; ++j) {
            s[0][j] = __expf(s[0][j] - mrow[j]);
            s[1][j] = __expf(s[1][j] - mrow[j]);
            float ps = s[0][j] + s[1][j];
            ps += __shfl_xor(ps, 1, 32);
            ps += __shfl_xor(ps, 2, 32);
            ps += __shfl_xor(ps, 4, 32);
            ps += __shfl_xor(ps, 8, 32);
            lrow[j] = lrow[j] * alpha[j] + ps;
        }
#pragma unroll
        for (int dt = 0; dt < 16; ++dt)
#pragma unroll
            for (int j = 0; j < 8; ++j) o[dt][j] *= alpha[j];

        // ---- P: C-layout -> LDS -> A-layout (wave-local transpose) ----
#pragma unroll
        for (int nt = 0; nt < 2; ++nt)
#pragma unroll
            for (int j = 0; j < 8; ++j)
                Pld[wave][j + 8 * hi][nt * 16 + r] = (__bf16)s[nt][j];
        asm volatile("s_wait_dscnt 0" ::: "memory");  // wave-ordered DS: stores visible
        v16bf pf = cat8(*(const v8bf*)&Pld[wave][r][hi * 8],
                        *(const v8bf*)&Pld[wave][r][16 + hi * 8]);
        // ---- O += P V  (V^T gives contiguous key-major B fragments) ----
#pragma unroll
        for (int dt = 0; dt < 16; ++dt) {
            const __bf16* vp = vT + (size_t)(dt * 16 + r) * T_TOK + kb0 + hi * 8;
            v16bf vf = cat8(*(const v8bf*)(vp), *(const v8bf*)(vp + 16));
            o[dt] = wmma_bf16(pf, vf, o[dt]);
        }
        asm volatile("" ::: "memory");
    }

    // ---- epilogue: divide by l, write bf16 ----
#pragma unroll
    for (int j = 0; j < 8; ++j) lrow[j] = 1.0f / lrow[j];
#pragma unroll
    for (int dt = 0; dt < 16; ++dt)
#pragma unroll
        for (int j = 0; j < 8; ++j)
            attn[(size_t)(qw0 + j + 8 * hi) * Q_SIZE + head * HEAD_DIM + dt * 16 + r] =
                (__bf16)(o[dt][j] * lrow[j]);
}

// ---------------------------------------------------------------------------
extern "C" void kernel_launch(void* const* d_in, const int* in_sizes, int n_in,
                              void* d_out, int out_size, void* d_ws, size_t ws_size,
                              hipStream_t stream) {
    const int*   positions = (const int*)d_in[0];
    const float* hidden    = (const float*)d_in[1];
    const float* Wqkv      = (const float*)d_in[2];
    const float* Wo        = (const float*)d_in[3];
    float*       out       = (float*)d_out;

    // Workspace layout (~59 MB):
    //   qkv fp32      T*QKV_N            (reused as bf16 attn output after rope)
    //   hidden bf16   T*HIDDEN
    //   Wqkv  bf16    QKV_N*HIDDEN
    //   Wo    bf16    HIDDEN*Q_SIZE
    //   q     bf16    T*Q_SIZE
    //   k     bf16    T*KV_SIZE
    //   vT    bf16    KV_SIZE*T
    float*  qkv_ws  = (float*)d_ws;
    __bf16* hid_bf  = (__bf16*)(qkv_ws + (size_t)T_TOK * QKV_N);
    __bf16* wqkv_bf = hid_bf + (size_t)T_TOK * HIDDEN;
    __bf16* wo_bf   = wqkv_bf + (size_t)QKV_N * HIDDEN;
    __bf16* qb      = wo_bf + (size_t)HIDDEN * Q_SIZE;
    __bf16* kbuf    = qb + (size_t)T_TOK * Q_SIZE;
    __bf16* vTb     = kbuf + (size_t)T_TOK * KV_SIZE;
    __bf16* attn_bf = (__bf16*)qkv_ws;  // qkv fp32 dead after rope_pack

    const size_t n_hid  = (size_t)T_TOK * HIDDEN;
    const size_t n_wqkv = (size_t)QKV_N * HIDDEN;
    const size_t n_wo   = (size_t)HIDDEN * Q_SIZE;

    cvt_bf16_kernel<<<dim3(n_hid  / 2048), dim3(256), 0, stream>>>(hidden, hid_bf);
    cvt_bf16_kernel<<<dim3(n_wqkv / 2048), dim3(256), 0, stream>>>(Wqkv, wqkv_bf);
    cvt_bf16_kernel<<<dim3(n_wo   / 2048), dim3(256), 0, stream>>>(Wo, wo_bf);

    gemm_nt_kernel<<<dim3(QKV_N / 32, T_TOK / 256), dim3(256), 0, stream>>>(
        hid_bf, wqkv_bf, qkv_ws, T_TOK, QKV_N, HIDDEN);

    rope_pack_kernel<<<dim3(T_TOK), dim3(256), 0, stream>>>(
        positions, qkv_ws, qb, kbuf, vTb);

    attn_fa_kernel<<<dim3(T_TOK / 64, NUM_HEADS), dim3(128), 0, stream>>>(
        qb, kbuf, vTb, attn_bf);

    // out[T,HIDDEN] = attn_bf16[T,Q_SIZE] * Wo_bf16[HIDDEN,Q_SIZE]^T
    // NOTE: attn_bf aliases qkv_ws; A is fully written before this launch.
    gemm_nt_kernel<<<dim3(HIDDEN / 32, T_TOK / 256), dim3(256), 0, stream>>>(
        attn_bf, wo_bf, out, T_TOK, HIDDEN, Q_SIZE);
}
